// LDGCNN_2628519985496
// MI455X (gfx1250) — compile-verified, matrix operand order checked by reading, not run on hardware
//
#include <hip/hip_runtime.h>
#include <hip/hip_bf16.h>

typedef __attribute__((ext_vector_type(16))) _Float16 v16h;
typedef __attribute__((ext_vector_type(8)))  float    v8f;

#define NEG 0.2f
#define KNB 20
#define NPTS 2048
#define BATCH 8
#define MAXKPAD 416   // max padded 2C (block4: 2*195 -> 416)
#define MAXCPAD 352   // max padded C (block5 input: 323 -> 352)

__device__ __forceinline__ float leaky(float x) { return x > 0.f ? x : NEG * x; }

__device__ __forceinline__ void atomicMaxF(float* addr, float val) {
    int* ai = (int*)addr;
    int old = *ai;
    while (__int_as_float(old) < val) {
        int assumed = old;
        old = atomicCAS(ai, assumed, __float_as_int(val));
        if (old == assumed) break;
    }
}

// map channel-within-chunk (0..31) -> (half, j) of the 16-bit WMMA operand layout
// forward map: c32 = ((j<8)? j : j+8) + half*8
__device__ __forceinline__ void chmap(int c32, int& half, int& j) {
    if (c32 < 16) { half = c32 >> 3; j = c32 & 7; }
    else          { int cx = c32 - 16; half = cx >> 3; j = 8 + (cx & 7); }
}

// ---------------- transpose x (B,3,N) -> pts (B,N,3) ----------------
__global__ void k_transpose(const float* __restrict__ x, float* __restrict__ pts, int N) {
    int t = blockIdx.x * blockDim.x + threadIdx.x;
    if (t >= BATCH * N) return;
    int b = t / N, n = t % N;
    for (int c = 0; c < 3; ++c)
        pts[(size_t)t * 3 + c] = x[((size_t)b * 3 + c) * N + n];
}

__global__ void k_fill(float* __restrict__ p, float v, int n) {
    int t = blockIdx.x * blockDim.x + threadIdx.x;
    if (t < n) p[t] = v;
}

// ---- W (Cout,Cin) f32 -> swizzled f16 B-operand tiles: [(ot*nchunks+chunk)*32+lane]*16+j ----
__global__ void k_convert_w_sw(const float* __restrict__ W, _Float16* __restrict__ Wsw,
                               int Cin, int Kpad, int Cout) {
    int t = blockIdx.x * blockDim.x + threadIdx.x;
    int total = Cout * Kpad;
    if (t >= total) return;
    int o = t / Kpad, kk = t % Kpad;
    float v = (kk < Cin) ? W[(size_t)o * Cin + kk] : 0.f;
    int ot = o >> 4, l = o & 15;
    int chunk = kk >> 5, c32 = kk & 31;
    int half, j; chmap(c32, half, j);
    int lane = l + half * 16;
    int nchunks = Kpad >> 5;
    Wsw[(((size_t)ot * nchunks + chunk) * 32 + lane) * 16 + j] = (_Float16)v;
}

// ------- concat up to 5 f32 sources (row-major BN x ci) -> f16 (BN x Cpad), zero pad -------
__global__ void k_concat16(const float* s0, int c0, const float* s1, int c1,
                           const float* s2, int c2, const float* s3, int c3,
                           const float* s4, int c4,
                           _Float16* __restrict__ out, int Cpad, int rows) {
    int t = blockIdx.x * blockDim.x + threadIdx.x;
    if (t >= rows * Cpad) return;
    size_t row = (size_t)t / Cpad;
    int ch = t % Cpad;
    float v = 0.f;
    int c = ch;
    if (c < c0) v = s0[row * c0 + c];
    else { c -= c0;
      if (c < c1) v = s1[row * c1 + c];
      else { c -= c1;
        if (c < c2) v = s2[row * c2 + c];
        else { c -= c2;
          if (c < c3) v = s3[row * c3 + c];
          else { c -= c3;
            if (c < c4) v = s4[row * c4 + c];
          }
        }
      }
    }
    out[(size_t)row * Cpad + ch] = (_Float16)v;
}

// ---------------- squared norms per row ----------------
__global__ void k_sqn(const _Float16* __restrict__ feat, float* __restrict__ sqn,
                      int Cpad, int rows) {
    int t = blockIdx.x * blockDim.x + threadIdx.x;
    if (t >= rows) return;
    const _Float16* r = feat + (size_t)t * Cpad;
    float s = 0.f;
    for (int c = 0; c < Cpad; ++c) { float v = (float)r[c]; s += v * v; }
    sqn[t] = s;
}

// ---------------- kNN top-20 (smallest dist, incl. self), LDS candidate tiles ----------------
#define KNN_CHUNK 32
__global__ void k_knn(const _Float16* __restrict__ feat, const float* __restrict__ sqn,
                      int* __restrict__ idx, int N, int Cpad) {
    __shared__ float scand[KNN_CHUNK * 224];
    __shared__ float ssq[KNN_CHUNK];
    int b = blockIdx.y;
    int q = blockIdx.x * blockDim.x + threadIdx.x;   // query point
    const _Float16* qrow = feat + ((size_t)b * N + q) * Cpad;
    float qs = sqn[b * N + q];
    float best[KNB]; int bidx[KNB];
    for (int i = 0; i < KNB; ++i) { best[i] = 1e30f; bidx[i] = 0; }
    for (int m0 = 0; m0 < N; m0 += KNN_CHUNK) {
        __syncthreads();
        int tot = KNN_CHUNK * Cpad;
        for (int i = threadIdx.x; i < tot; i += blockDim.x) {
            int mm = i / Cpad, ch = i % Cpad;
            scand[mm * Cpad + ch] = (float)feat[((size_t)b * N + m0 + mm) * Cpad + ch];
        }
        if (threadIdx.x < KNN_CHUNK) ssq[threadIdx.x] = sqn[b * N + m0 + threadIdx.x];
        __syncthreads();
        for (int mm = 0; mm < KNN_CHUNK; ++mm) {
            float d = 0.f;
            const float* crow = &scand[mm * Cpad];
            for (int ch = 0; ch < Cpad; ++ch) d += (float)qrow[ch] * crow[ch];
            float dist = qs + ssq[mm] - 2.f * d;
            if (dist < 0.f) dist = 0.f;
            if (dist < best[KNB - 1]) {
                int pos = KNB - 1;
                while (pos > 0 && best[pos - 1] > dist) {
                    best[pos] = best[pos - 1]; bidx[pos] = bidx[pos - 1]; --pos;
                }
                best[pos] = dist; bidx[pos] = m0 + mm;
            }
        }
    }
    for (int i = 0; i < KNB; ++i)
        idx[((size_t)b * N + q) * KNB + i] = bidx[i];
}

// -------- EdgeConv: block = (b, 16-pt tile); waves = Cout tiles; LDS-staged A tiles --------
__global__ void k_edgeconv(const _Float16* __restrict__ feat, const int* __restrict__ idx,
                           const _Float16* __restrict__ Wsw, const float* __restrict__ sc,
                           const float* __restrict__ bi, float* __restrict__ out,
                           int N, int Cpad, int Cin, int Kpad, int Cout) {
    __shared__ __align__(32) _Float16 sA[16 * MAXKPAD];  // A tiles, swizzled operand layout
    __shared__ int snbr[16];
    const int nchunks = Kpad >> 5;
    int t = blockIdx.x;
    int ntiles = N >> 4;
    int nt = t % ntiles, b = t / ntiles;
    int n0 = nt * 16;
    int lane = threadIdx.x;            // 0..31
    int ot = threadIdx.y;              // Cout tile for this wave
    int l = lane & 15, half = lane >> 4;
    int o0 = ot * 16;
    int tid = ot * 32 + lane;
    int nthr = blockDim.y * 32;

    float ss = sc[o0 + l], bb = bi[o0 + l];
    const v16h* sAv = (const v16h*)sA;
    const v16h* Bv  = (const v16h*)Wsw;
    v8f accmax;
#pragma unroll
    for (int r = 0; r < 8; ++r) accmax[r] = -1e30f;

    for (int k = 0; k < KNB; ++k) {
        __syncthreads();
        if (tid < 16) snbr[tid] = idx[((size_t)b * N + n0 + tid) * KNB + k];
        __syncthreads();
        // stage edge features [nbr-ctr ; ctr ; 0] into swizzled A layout
        int tot = 16 * Kpad;
        for (int e = tid; e < tot; e += nthr) {
            int pl = e / Kpad, ch = e - pl * Kpad;
            const _Float16* crow = feat + ((size_t)b * N + n0 + pl) * Cpad;
            float v;
            if (ch < Cin) {
                const _Float16* nrow = feat + ((size_t)b * N + snbr[pl]) * Cpad;
                v = (float)nrow[ch] - (float)crow[ch];
            } else if (ch < 2 * Cin) {
                v = (float)crow[ch - Cin];
            } else v = 0.f;
            int chunk = ch >> 5, c32 = ch & 31;
            int hh, j; chmap(c32, hh, j);
            sA[((chunk * 32 + pl + hh * 16) << 4) + j] = (_Float16)v;
        }
        __syncthreads();
        // one (16pt x 16co) tile: A from LDS (32B/lane), B from global (32B/lane)
        v8f c = {};
        for (int ch = 0; ch < nchunks; ++ch) {
            v16h a  = sAv[ch * 32 + lane];
            v16h bm = Bv[((size_t)ot * nchunks + ch) * 32 + lane];
            c = __builtin_amdgcn_wmma_f32_16x16x32_f16(false, a, false, bm,
                                                       (short)0, c, false, false);
        }
#pragma unroll
        for (int r = 0; r < 8; ++r)
            accmax[r] = fmaxf(accmax[r], leaky(c[r] * ss + bb));
    }
    // D layout: col = o0+l, row(point) = n0 + r + half*8
#pragma unroll
    for (int r = 0; r < 8; ++r)
        out[((size_t)b * N + n0 + r + half * 8) * Cout + o0 + l] = accmax[r];
}

// -------- block5: pointwise conv 323->1024 + affine + leaky + global max pool --------
__global__ void k_conv5(const _Float16* __restrict__ feat, const _Float16* __restrict__ Wsw,
                        const float* __restrict__ sc, const float* __restrict__ bi,
                        float* __restrict__ gfeat, int N, int Cpad, int Cout) {
    __shared__ __align__(32) _Float16 sA[16 * MAXCPAD];
    const int nchunks = Cpad >> 5;
    int ntiles = N >> 4;
    int ogroups = (Cout >> 4) / 8;      // 8 waves per block, one Cout tile each
    int t = blockIdx.x;
    int og = t % ogroups; t /= ogroups;
    int nt = t % ntiles;  int b = t / ntiles;
    int n0 = nt * 16;
    int lane = threadIdx.x;
    int ot = og * 8 + threadIdx.y;
    int l = lane & 15, half = lane >> 4;
    int o0 = ot * 16;
    int tid = threadIdx.y * 32 + lane, nthr = blockDim.y * 32;

    // stage activation tile once (pure f16 copy into swizzled operand layout)
    int tot = 16 * Cpad;
    for (int e = tid; e < tot; e += nthr) {
        int pl = e / Cpad, ch = e - pl * Cpad;
        int chunk = ch >> 5, c32 = ch & 31;
        int hh, j; chmap(c32, hh, j);
        sA[((chunk * 32 + pl + hh * 16) << 4) + j] =
            feat[((size_t)b * N + n0 + pl) * Cpad + ch];
    }
    __syncthreads();

    const v16h* sAv = (const v16h*)sA;
    const v16h* Bv  = (const v16h*)Wsw;
    float ss = sc[o0 + l], bb = bi[o0 + l];
    v8f c = {};
    for (int ch = 0; ch < nchunks; ++ch) {
        v16h a  = sAv[ch * 32 + lane];
        v16h bm = Bv[((size_t)ot * nchunks + ch) * 32 + lane];
        c = __builtin_amdgcn_wmma_f32_16x16x32_f16(false, a, false, bm,
                                                   (short)0, c, false, false);
    }
#pragma unroll
    for (int r = 0; r < 8; ++r)
        atomicMaxF(&gfeat[(size_t)b * Cout + o0 + l], leaky(c[r] * ss + bb));
}

// ---------------- FC head ----------------
__global__ void k_fc(const float* __restrict__ in, const float* __restrict__ W,
                     const float* __restrict__ bias, const float* __restrict__ scale,
                     const float* __restrict__ beta, float* __restrict__ out,
                     int Cin, int Cout, int useAffine) {
    int o = blockIdx.x * blockDim.x + threadIdx.x;
    int b = blockIdx.y;
    if (o >= Cout) return;
    const float* row = in + (size_t)b * Cin;
    float acc = bias[o];
    for (int c = 0; c < Cin; ++c) acc += row[c] * W[(size_t)o * Cin + c];
    if (useAffine) acc = leaky(acc * scale[o] + beta[o]);
    out[(size_t)b * Cout + o] = acc;
}

__global__ void k_copy(const float* __restrict__ src, float* __restrict__ dst, int n) {
    int t = blockIdx.x * blockDim.x + threadIdx.x;
    if (t < n) dst[t] = src[t];
}

// ================================================================
extern "C" void kernel_launch(void* const* d_in, const int* in_sizes, int n_in,
                              void* d_out, int out_size, void* d_ws, size_t ws_size,
                              hipStream_t stream) {
    const int B = BATCH, N = NPTS;
    const float* x = (const float*)d_in[0];
    const float* W[5]  = { (const float*)d_in[1], (const float*)d_in[4], (const float*)d_in[7],
                           (const float*)d_in[10], (const float*)d_in[13] };
    const float* S[5]  = { (const float*)d_in[2], (const float*)d_in[5], (const float*)d_in[8],
                           (const float*)d_in[11], (const float*)d_in[14] };
    const float* Bb[5] = { (const float*)d_in[3], (const float*)d_in[6], (const float*)d_in[9],
                           (const float*)d_in[12], (const float*)d_in[15] };
    const float* fW1 = (const float*)d_in[16]; const float* fb1 = (const float*)d_in[17];
    const float* fs1 = (const float*)d_in[18]; const float* fB1 = (const float*)d_in[19];
    const float* fW2 = (const float*)d_in[20]; const float* fb2 = (const float*)d_in[21];
    const float* fs2 = (const float*)d_in[22]; const float* fB2 = (const float*)d_in[23];
    const float* fW3 = (const float*)d_in[24]; const float* fb3 = (const float*)d_in[25];
    float* out = (float*)d_out;

    // ---- workspace carve (all buffers fully rewritten every call) ----
    char* wsp = (char*)d_ws;
    size_t off = 0;
    auto carve = [&](size_t bytes) -> void* {
        void* p = wsp + off;
        off += (bytes + 255) & ~(size_t)255;
        return p;
    };
    const size_t BN = (size_t)B * N;
    float*     pts    = (float*)carve(BN * 3 * 4);
    float*     f1     = (float*)carve(BN * 64 * 4);
    float*     f2     = (float*)carve(BN * 64 * 4);
    float*     f3     = (float*)carve(BN * 64 * 4);
    float*     f4     = (float*)carve(BN * 128 * 4);
    _Float16*  feat16 = (_Float16*)carve(BN * MAXCPAD * 2);
    float*     sqn    = (float*)carve(BN * 4);
    int*       idx    = (int*)carve(BN * KNB * 4);
    _Float16*  Wsw    = (_Float16*)carve((size_t)1024 * MAXCPAD * 2);
    float*     gfeat  = (float*)carve((size_t)B * 1024 * 4);
    float*     h1     = (float*)carve((size_t)B * 512 * 4);
    float*     h2     = (float*)carve((size_t)B * 256 * 4);
    (void)ws_size; (void)in_sizes; (void)n_in; (void)out_size;

    k_transpose<<<(B * N + 255) / 256, 256, 0, stream>>>(x, pts, N);

    // ---- four EdgeConv blocks ----
    struct Blk { int Cin, Cpad, Kpad, Cout; float* dst; };
    Blk blks[4] = {
        {   3,  32,  32,  64, f1 },
        {  67,  96, 160,  64, f2 },
        { 131, 160, 288,  64, f3 },
        { 195, 224, 416, 128, f4 },
    };
    for (int i = 0; i < 4; ++i) {
        const Blk& bk = blks[i];
        int c1 = (i >= 1) ? 64 : 0, c2 = (i >= 2) ? 64 : 0, c3 = (i >= 3) ? 64 : 0;
        size_t celts = BN * bk.Cpad;
        k_concat16<<<(celts + 255) / 256, 256, 0, stream>>>(
            pts, 3, f1, c1, f2, c2, f3, c3, (const float*)nullptr, 0,
            feat16, bk.Cpad, (int)BN);
        k_sqn<<<((int)BN + 255) / 256, 256, 0, stream>>>(feat16, sqn, bk.Cpad, (int)BN);
        dim3 kg(N / 128, B);
        k_knn<<<kg, 128, 0, stream>>>(feat16, sqn, idx, N, bk.Cpad);
        int welts = bk.Cout * bk.Kpad;
        k_convert_w_sw<<<(welts + 255) / 256, 256, 0, stream>>>(W[i], Wsw, 2 * bk.Cin,
                                                               bk.Kpad, bk.Cout);
        dim3 eb(32, bk.Cout / 16);       // one wave per Cout tile, shared staged A
        k_edgeconv<<<B * (N / 16), eb, 0, stream>>>(feat16, idx, Wsw, S[i], Bb[i], bk.dst,
                                                    N, bk.Cpad, bk.Cin, bk.Kpad, bk.Cout);
    }

    // ---- block 5: pointwise conv 323 -> 1024, global max pool ----
    {
        const int Cin = 323, Cpad = MAXCPAD, Cout = 1024;
        size_t celts = BN * Cpad;
        k_concat16<<<(celts + 255) / 256, 256, 0, stream>>>(
            pts, 3, f1, 64, f2, 64, f3, 64, f4, 128, feat16, Cpad, (int)BN);
        int welts = Cout * Cpad;
        k_convert_w_sw<<<(welts + 255) / 256, 256, 0, stream>>>(W[4], Wsw, Cin, Cpad, Cout);
        k_fill<<<(B * Cout + 255) / 256, 256, 0, stream>>>(gfeat, -1e30f, B * Cout);
        dim3 cb(32, 8);                  // 8 waves = 8 Cout tiles per block
        int nblk = B * (N / 16) * ((Cout / 16) / 8);
        k_conv5<<<nblk, cb, 0, stream>>>(feat16, Wsw, S[4], Bb[4], gfeat, N, Cpad, Cout);
    }

    // ---- FC head ----
    {
        dim3 g1((512 + 127) / 128, B);
        k_fc<<<g1, 128, 0, stream>>>(gfeat, fW1, fb1, fs1, fB1, h1, 1024, 512, 1);
        dim3 g2((256 + 127) / 128, B);
        k_fc<<<g2, 128, 0, stream>>>(h1, fW2, fb2, fs2, fB2, h2, 512, 256, 1);
        dim3 g3((40 + 127) / 128, B);
        k_fc<<<g3, 128, 0, stream>>>(h2, fW3, fb3, nullptr, nullptr, out, 256, 40, 0);
        k_copy<<<(B * 1024 + 255) / 256, 256, 0, stream>>>(gfeat, out + B * 40, B * 1024);
    }
}
